// CAM_56951266345585
// MI455X (gfx1250) — matrix operand order repaired and checked by reading, not compiled, not run on hardware
//
#include <hip/hip_runtime.h>

// ---------------- types / constants ----------------
typedef __bf16 bf16_t;
typedef __attribute__((ext_vector_type(16))) __bf16   v16bf;
typedef __attribute__((ext_vector_type(8)))  float    v8f;
typedef __attribute__((ext_vector_type(4)))  unsigned v4u;
typedef __attribute__((ext_vector_type(8)))  unsigned v8u;

#define BATCH   16
#define NTOK    3136
#define DIM     256
#define NHEADS  8
#define HD      32
#define KEEP    1568
#define HALF    784
#define RNUM    471     // merged sources
#define UNMN    313     // unmerged sources
#define MN      1097    // merged token count
#define TOPK    137     // MN/8
#define NPAD    1120    // MN padded to multiple of 32
#define MTILES  69      // ceil(MN/16)
#define SCALE_F 0.17677669529663687f  // HEAD_DIM^-0.5

// ---------------- WMMA fragment helpers (contiguous K-half order: k = 16*(lane>>4) + e) ----------------
// A-fragment: lane holds row m = lane&15; its 8 dwords (16 bf16) of a 32-wide
// K window are one contiguous 8-dword run (K half) selected by lane>>4.
static __device__ __forceinline__ v16bf frag_a_win(const unsigned* win, int lane) {
  int hgf = lane >> 4;
  v4u lo = *(const v4u*)(win + hgf * 8);
  v4u hi = *(const v4u*)(win + hgf * 8 + 4);
  v8u raw;
  raw[0] = lo[0]; raw[1] = lo[1]; raw[2] = lo[2]; raw[3] = lo[3];
  raw[4] = hi[0]; raw[5] = hi[1]; raw[6] = hi[2]; raw[7] = hi[3];
  return __builtin_bit_cast(v16bf, raw);
}
// B-fragment from K-contiguous storage: lane holds column n = lane&15,
// K half selected by lane>>4; 8 consecutive dwords.
static __device__ __forceinline__ v16bf frag_b_kcont(const unsigned* col, int lane) {
  v8u raw = *(const v8u*)(col + (lane >> 4) * 8);
  return __builtin_bit_cast(v16bf, raw);
}

static __device__ __forceinline__ unsigned fkey(float x) {
  unsigned u = __float_as_uint(x);
  return (u & 0x80000000u) ? ~u : (u | 0x80000000u);
}
static __device__ __forceinline__ int wave_sum_i(int v) {
#pragma unroll
  for (int o = 16; o > 0; o >>= 1) v += __shfl_xor(v, o, 32);
  return v;
}
static __device__ __forceinline__ float wave_sum_f(float v) {
#pragma unroll
  for (int o = 16; o > 0; o >>= 1) v += __shfl_xor(v, o, 32);
  return v;
}
static __device__ __forceinline__ float wave_max_f(float v) {
#pragma unroll
  for (int o = 16; o > 0; o >>= 1) v = fmaxf(v, __shfl_xor(v, o, 32));
  return v;
}

// ---------------- K0: f32 -> bf16 convert ----------------
__global__ void cvt_f32_bf16(const float* __restrict__ in, bf16_t* __restrict__ out, long long n) {
  long long i = (long long)blockIdx.x * blockDim.x + threadIdx.x;
  if (i < n) out[i] = (bf16_t)in[i];
}

// ---------------- K0b: pack weight (KxN f32) into per-lane B-fragment order ----------------
// Wp layout: [kt = K/32][ntile = N/16][lane 0..31][e 0..15], element e of lane
// is W[kt*32 + (lane>>4)*16 + e][ntile*16 + (lane&15)].
__global__ void pack_b_kernel(const float* __restrict__ W, bf16_t* __restrict__ Wp,
                              int K, int N) {
  long long i = (long long)blockIdx.x * blockDim.x + threadIdx.x;
  long long total = (long long)K * N;
  if (i >= total) return;
  int e = (int)(i & 15);
  long long t = i >> 4;
  int lane = (int)(t & 31);
  t >>= 5;
  int ntiles = N >> 4;
  int ntile = (int)(t % ntiles);
  int kt = (int)(t / ntiles);
  int n = ntile * 16 + (lane & 15);
  int k = kt * 32 + (lane >> 4) * 16 + e;
  Wp[i] = (bf16_t)W[(size_t)k * N + n];
}

// ---------------- K2: tvec = W_k @ (mean(x) @ W_gt) ----------------
__global__ void gt_tvec_kernel(const float* __restrict__ x, const float* __restrict__ Wgt,
                               const float* __restrict__ Wk, float* __restrict__ tvec) {
  int b = blockIdx.x, c = threadIdx.x;  // 256 threads
  __shared__ float xm[DIM];
  __shared__ float gt[DIM];
  const float* xb = x + (size_t)b * NTOK * DIM;
  float acc = 0.0f;
  for (int n = 0; n < NTOK; ++n) acc += xb[(size_t)n * DIM + c];
  xm[c] = acc * (1.0f / NTOK);
  __syncthreads();
  float g = 0.0f;
  for (int k = 0; k < DIM; ++k) g += xm[k] * Wgt[k * DIM + c];
  gt[c] = g;
  __syncthreads();
  float t = 0.0f;
  for (int k = 0; k < DIM; ++k) t += Wk[c * DIM + k] * gt[k];
  tvec[b * DIM + c] = t;
}

// ---------------- K3: score[b,n] = SCALE * dot(x[b,n], tvec[b]) ----------------
__global__ void score_kernel(const float* __restrict__ x, const float* __restrict__ tvec,
                             float* __restrict__ score) {
  int t = blockIdx.x * blockDim.x + threadIdx.x;
  if (t >= BATCH * NTOK) return;
  int b = t / NTOK;
  const float* xr = x + (size_t)t * DIM;
  const float* tv = tvec + b * DIM;
  float acc = 0.0f;
  for (int c = 0; c < DIM; ++c) acc += xr[c] * tv[c];
  score[t] = acc * SCALE_F;
}

// ---------------- K4: per-batch descending-stable sort of scores -> reserve ----------------
__global__ void sort_reserve_kernel(const float* __restrict__ score, int* __restrict__ reserve) {
  const int SZ = 4096;
  __shared__ float ks[SZ];
  __shared__ int   vs[SZ];
  int b = blockIdx.x, tid = threadIdx.x, nt = blockDim.x;
  for (int i = tid; i < SZ; i += nt) {
    if (i < NTOK) { ks[i] = score[b * NTOK + i]; vs[i] = i; }
    else          { ks[i] = -INFINITY;           vs[i] = 0x7FFFFFFF; }
  }
  __syncthreads();
  for (int k = 2; k <= SZ; k <<= 1) {
    for (int j = k >> 1; j > 0; j >>= 1) {
      for (int t = tid; t < SZ; t += nt) {
        int ixj = t ^ j;
        if (ixj > t) {
          float s1 = ks[t], s2 = ks[ixj];
          int   i1 = vs[t], i2 = vs[ixj];
          bool before12 = (s1 > s2) || (s1 == s2 && i1 < i2);
          bool up = ((t & k) == 0);
          bool sw = up ? !before12 : before12;
          if (sw) { ks[t] = s2; ks[ixj] = s1; vs[t] = i2; vs[ixj] = i1; }
        }
      }
      __syncthreads();
    }
  }
  for (int i = tid; i < KEEP; i += nt) reserve[b * KEEP + i] = vs[i];
}

// ---------------- generic bf16 WMMA GEMM: 16x64 per wave, direct b128 frag loads ----------------
static __device__ __forceinline__ void store_tile(v8f acc, float* Cf, bf16_t* Cb,
                                                  const float* bias, size_t baseC,
                                                  int M, int N, int m0, int n0, int lane) {
  int n = lane & 15, hgf = lane >> 4;
  float badd = bias ? bias[n0 + n] : 0.0f;
#pragma unroll
  for (int v = 0; v < 8; ++v) {
    int row = m0 + v + 8 * hgf;
    if (row < M) {
      size_t o = baseC + (size_t)row * N + n0 + n;
      if (Cf) Cf[o] = acc[v] + badd;
      else    Cb[o] = (bf16_t)acc[v];
    }
  }
}

__global__ void gemm_bf16_wmma(const bf16_t* __restrict__ A, long long sAb,
                               const int* __restrict__ rowidx, int ridStride,
                               const bf16_t* __restrict__ Bp,  // packed (pack_b_kernel)
                               const float* __restrict__ bias,
                               float* __restrict__ Cf, bf16_t* __restrict__ Cb, long long sCb,
                               int M, int N, int K) {
  int mt = blockIdx.x, ng = blockIdx.y, b = blockIdx.z;
  int lane = threadIdx.x;  // 32
  int m0 = mt * 16;
  int row = m0 + (lane & 15);
  if (row >= M) row = M - 1;
  int rid = rowidx ? rowidx[(size_t)b * ridStride + row] : row;
  const unsigned* arow = (const unsigned*)(A + (size_t)b * sAb + (size_t)rid * K);
  int ntiles = N >> 4;
  int nt0 = ng * 4;
  v8f a0 = {}, a1 = {}, a2 = {}, a3 = {};
  for (int kt = 0; kt < (K >> 5); ++kt) {
    v16bf af = frag_a_win(arow + kt * 16, lane);
    const unsigned* bb = (const unsigned*)Bp + (((size_t)kt * ntiles + nt0) * 32 + lane) * 8;
    __builtin_prefetch((const void*)(bb + (size_t)ntiles * 256), 0, 1);
    v16bf b0 = __builtin_bit_cast(v16bf, *(const v8u*)(bb));
    v16bf b1 = __builtin_bit_cast(v16bf, *(const v8u*)(bb + 32 * 8));
    v16bf b2 = __builtin_bit_cast(v16bf, *(const v8u*)(bb + 64 * 8));
    v16bf b3 = __builtin_bit_cast(v16bf, *(const v8u*)(bb + 96 * 8));
    a0 = __builtin_amdgcn_wmma_f32_16x16x32_bf16(false, af, false, b0, (short)0, a0, false, false);
    a1 = __builtin_amdgcn_wmma_f32_16x16x32_bf16(false, af, false, b1, (short)0, a1, false, false);
    a2 = __builtin_amdgcn_wmma_f32_16x16x32_bf16(false, af, false, b2, (short)0, a2, false, false);
    a3 = __builtin_amdgcn_wmma_f32_16x16x32_bf16(false, af, false, b3, (short)0, a3, false, false);
  }
  size_t baseC = (size_t)b * sCb;
  store_tile(a0, Cf, Cb, bias, baseC, M, N, m0, (nt0 + 0) * 16, lane);
  store_tile(a1, Cf, Cb, bias, baseC, M, N, m0, (nt0 + 1) * 16, lane);
  store_tile(a2, Cf, Cb, bias, baseC, M, N, m0, (nt0 + 2) * 16, lane);
  store_tile(a3, Cf, Cb, bias, baseC, M, N, m0, (nt0 + 3) * 16, lane);
}

// ---------------- K5b: row L2-normalize metric (in place, bf16) ----------------
__global__ void normalize_metric(bf16_t* __restrict__ metric) {
  int row = blockIdx.x, b = blockIdx.y, tid = threadIdx.x;  // 256 threads
  __shared__ float red[DIM];
  bf16_t* m = metric + ((size_t)b * KEEP + row) * DIM;
  float v = (float)m[tid];
  red[tid] = v * v;
  __syncthreads();
  for (int s = DIM / 2; s > 0; s >>= 1) {
    if (tid < s) red[tid] += red[tid + s];
    __syncthreads();
  }
  float inv = 1.0f / sqrtf(red[0]);
  m[tid] = (bf16_t)(v * inv);
}

// ---------------- K6: sim = a @ bm^T, per-row max + first-argmax ----------------
// a-row m -> metric row 2m (K-contiguous); bm-col n -> metric row 2n+1 (K-contiguous).
__global__ void sim_rowmax(const bf16_t* __restrict__ metric, float* __restrict__ max_sim,
                           int* __restrict__ dst_all) {
  __shared__ float Ct[16 * 16];
  __shared__ float bestv[16];
  __shared__ int   besti[16];
  int mt = blockIdx.x, b = blockIdx.y, lane = threadIdx.x;  // 32 threads
  int m0 = mt * 16;
  const bf16_t* mb = metric + (size_t)b * KEEP * DIM;
  const unsigned* arow = (const unsigned*)(mb + (size_t)(2 * (m0 + (lane & 15))) * DIM);
  if (lane < 16) { bestv[lane] = -INFINITY; besti[lane] = 0; }
  __syncthreads();
  int n = lane & 15, hgf = lane >> 4;
  for (int nt = 0; nt < HALF / 16; ++nt) {
    int n0 = nt * 16;
    const unsigned* bcol = (const unsigned*)(mb + (size_t)(2 * (n0 + n) + 1) * DIM);
    v8f acc = {};
#pragma unroll
    for (int kc = 0; kc < 8; ++kc) {
      v16bf af  = frag_a_win(arow + kc * 16, lane);
      v16bf bfr = frag_b_kcont(bcol + kc * 16, lane);
      acc = __builtin_amdgcn_wmma_f32_16x16x32_bf16(false, af, false, bfr, (short)0, acc, false, false);
    }
#pragma unroll
    for (int v = 0; v < 8; ++v) Ct[(v + 8 * hgf) * 16 + n] = acc[v];
    __syncthreads();
    if (lane < 16) {
      float bv = bestv[lane]; int bi = besti[lane];
      for (int j = 0; j < 16; ++j) {
        float val = Ct[lane * 16 + j];
        if (val > bv) { bv = val; bi = n0 + j; }
      }
      bestv[lane] = bv; besti[lane] = bi;
    }
    __syncthreads();
  }
  if (lane < 16) {
    max_sim[b * HALF + m0 + lane] = bestv[lane];
    dst_all[b * HALF + m0 + lane] = besti[lane];
  }
}

// ---------------- K7: sort max_sim desc -> src/unm/dst idx, inverse map, counts ----------------
__global__ void sort_rank_kernel(const float* __restrict__ max_sim, const int* __restrict__ dst_all,
                                 int* __restrict__ src_idx, int* __restrict__ unm_idx,
                                 int* __restrict__ dst_idx, int* __restrict__ invrow,
                                 int* __restrict__ counts) {
  const int SZ = 1024;
  __shared__ float ks[SZ];
  __shared__ int   vs[SZ];
  __shared__ int   cnt[HALF];
  int b = blockIdx.x, tid = threadIdx.x, nt = blockDim.x;  // 256 threads
  for (int i = tid; i < SZ; i += nt) {
    if (i < HALF) { ks[i] = max_sim[b * HALF + i]; vs[i] = i; }
    else          { ks[i] = -INFINITY;             vs[i] = 0x7FFFFFFF; }
  }
  for (int i = tid; i < HALF; i += nt) cnt[i] = 0;
  __syncthreads();
  for (int k = 2; k <= SZ; k <<= 1) {
    for (int j = k >> 1; j > 0; j >>= 1) {
      for (int t = tid; t < SZ; t += nt) {
        int ixj = t ^ j;
        if (ixj > t) {
          float s1 = ks[t], s2 = ks[ixj];
          int   i1 = vs[t], i2 = vs[ixj];
          bool before12 = (s1 > s2) || (s1 == s2 && i1 < i2);
          bool up = ((t & k) == 0);
          bool sw = up ? !before12 : before12;
          if (sw) { ks[t] = s2; ks[ixj] = s1; vs[t] = i2; vs[ixj] = i1; }
        }
      }
      __syncthreads();
    }
  }
  for (int t = tid; t < HALF; t += nt) {
    int s = vs[t];
    if (t < RNUM) {
      src_idx[b * RNUM + t] = s;
      int d = dst_all[b * HALF + s];
      dst_idx[b * RNUM + t] = d;
      invrow[b * HALF + s] = UNMN + d;
      atomicAdd(&cnt[d], 1);
    } else {
      int i = t - RNUM;
      unm_idx[b * UNMN + i] = s;
      invrow[b * HALF + s] = i;
    }
  }
  __syncthreads();
  for (int d = tid; d < HALF; d += nt) counts[b * HALF + d] = cnt[d];
}

// ---------------- K8: build x_merge (unmerged gather + merged mean) ----------------
__global__ void build_xmerge(const float* __restrict__ x, const int* __restrict__ reserve,
                             const int* __restrict__ unm_idx, const int* __restrict__ src_idx,
                             const int* __restrict__ dst_idx, const int* __restrict__ counts,
                             bf16_t* __restrict__ xmerge) {
  __shared__ int di[RNUM];
  __shared__ int st[RNUM];
  int row = blockIdx.x, b = blockIdx.y, c = threadIdx.x;  // 256 threads
  if (row < UNMN) {
    int tok = reserve[b * KEEP + 2 * unm_idx[b * UNMN + row]];
    xmerge[((size_t)b * MN + row) * DIM + c] = (bf16_t)x[((size_t)b * NTOK + tok) * DIM + c];
  } else {
    int d = row - UNMN;
    for (int i = threadIdx.x; i < RNUM; i += blockDim.x) {
      di[i] = dst_idx[b * RNUM + i];
      st[i] = reserve[b * KEEP + 2 * src_idx[b * RNUM + i]];
    }
    __syncthreads();
    int tokd = reserve[b * KEEP + 2 * d + 1];
    float acc = x[((size_t)b * NTOK + tokd) * DIM + c];
    for (int i = 0; i < RNUM; ++i)
      if (di[i] == d) acc += x[((size_t)b * NTOK + st[i]) * DIM + c];
    acc /= (1.0f + (float)counts[b * HALF + d]);
    xmerge[((size_t)b * MN + row) * DIM + c] = (bf16_t)acc;
  }
}

// ---------------- K9b: V transpose with zero pad: Vt[b][d][k] = V[b][k][d] ----------------
__global__ void transpose_v(const bf16_t* __restrict__ qkv, bf16_t* __restrict__ Vt) {
  long long i = (long long)blockIdx.x * blockDim.x + threadIdx.x;
  long long total = (long long)BATCH * DIM * NPAD;
  if (i >= total) return;
  int k = (int)(i % NPAD);
  long long t = i / NPAD;
  int d = (int)(t % DIM);
  int b = (int)(t / DIM);
  bf16_t v = (bf16_t)0.0f;
  if (k < MN) v = qkv[((size_t)b * MN + k) * (3 * DIM) + 2 * DIM + d];
  Vt[i] = v;
}

// ---------------- K10: attention with exact per-row top-k mask ----------------
__global__ void attention_kernel(const bf16_t* __restrict__ qkv, const bf16_t* __restrict__ Vt,
                                 bf16_t* __restrict__ xattn) {
  extern __shared__ char smem[];
  float*  S = (float*)smem;                     // 16 x NPAD f32
  bf16_t* P = (bf16_t*)(smem + 16 * NPAD * 4);  // 16 x NPAD bf16
  int mt = blockIdx.x, h = blockIdx.y, b = blockIdx.z;
  int lane = threadIdx.x;  // 32
  int m0 = mt * 16;
  const bf16_t* base = qkv + (size_t)b * MN * (3 * DIM);
  int n = lane & 15, hgf = lane >> 4;
  // ---- Q fragment straight from global (K = 32 = full head dim) ----
  int qrow = m0 + n;
  if (qrow >= MN) qrow = MN - 1;
  v16bf qf = frag_a_win((const unsigned*)(base + (size_t)qrow * (3 * DIM) + h * HD), lane);
  // ---- S = scale * Q K^T ----
  for (int nt = 0; nt < NPAD / 16; ++nt) {
    int col0 = nt * 16;
    int krow = col0 + n;
    if (krow >= MN) krow = MN - 1;
    v16bf kf = frag_b_kcont((const unsigned*)(base + (size_t)krow * (3 * DIM) + DIM + h * HD), lane);
    v8f c = {};
    c = __builtin_amdgcn_wmma_f32_16x16x32_bf16(false, qf, false, kf, (short)0, c, false, false);
#pragma unroll
    for (int v = 0; v < 8; ++v) {
      int m = v + 8 * hgf;
      int col = col0 + n;
      S[m * NPAD + col] = (col < MN) ? c[v] * SCALE_F : -INFINITY;
    }
  }
  __syncthreads();
  // ---- per-row: exact kth-largest threshold (bitwise radix select), masked softmax ----
  for (int r = 0; r < 16; ++r) {
    const float* Sr = S + r * NPAD;
    unsigned T = 0;
    for (int bit = 31; bit >= 0; --bit) {
      unsigned cand = T | (1u << bit);
      int c = 0;
      for (int j = lane; j < MN; j += 32)
        if (fkey(Sr[j]) >= cand) ++c;
      if (wave_sum_i(c) >= TOPK) T = cand;
    }
    float mx = -INFINITY;
    for (int j = lane; j < MN; j += 32) mx = fmaxf(mx, Sr[j]);
    mx = wave_max_f(mx);
    float sum = 0.0f;
    for (int j = lane; j < MN; j += 32) {
      float s = Sr[j];
      if (fkey(s) >= T) sum += __expf(s - mx);
    }
    sum = wave_sum_f(sum);
    float inv = 1.0f / sum;
    bf16_t* Pr = P + r * NPAD;
    for (int j = lane; j < NPAD; j += 32) {
      float p = 0.0f;
      if (j < MN) {
        float s = Sr[j];
        if (fkey(s) >= T) p = __expf(s - mx) * inv;
      }
      Pr[j] = (bf16_t)p;
    }
  }
  __syncthreads();
  // ---- out tile = P @ V (16 x 32), V fragments from K-contiguous Vt ----
  const bf16_t* vtb = Vt + (size_t)(b * DIM + h * HD) * NPAD;
  for (int hn = 0; hn < 2; ++hn) {
    v8f acc = {};
    const bf16_t* vcol = vtb + (size_t)(hn * 16 + n) * NPAD;
    for (int kt = 0; kt < NPAD / 32; ++kt) {
      v16bf pf = frag_a_win((const unsigned*)(P + (size_t)n * NPAD + kt * 32), lane);
      v16bf vf = frag_b_kcont((const unsigned*)(vcol + kt * 32), lane);
      acc = __builtin_amdgcn_wmma_f32_16x16x32_bf16(false, pf, false, vf, (short)0, acc, false, false);
    }
#pragma unroll
    for (int v = 0; v < 8; ++v) {
      int row = m0 + v + 8 * hgf;
      if (row < MN)
        xattn[((size_t)b * MN + row) * DIM + h * HD + hn * 16 + n] = (bf16_t)acc[v];
    }
  }
}

// ---------------- K11: scatter restored rows into Y (bf16 image of x) ----------------
__global__ void scatter_rows(const bf16_t* __restrict__ xattn, const int* __restrict__ reserve,
                             const int* __restrict__ invrow, bf16_t* __restrict__ Y) {
  int j = blockIdx.x, b = blockIdx.y, c = threadIdx.x;  // 256 threads
  int row = (j & 1) ? (UNMN + (j >> 1)) : invrow[b * HALF + (j >> 1)];
  int tok = reserve[b * KEEP + j];
  Y[((size_t)b * NTOK + tok) * DIM + c] = xattn[((size_t)b * MN + row) * DIM + c];
}

// ---------------- host driver ----------------
extern "C" void kernel_launch(void* const* d_in, const int* in_sizes, int n_in,
                              void* d_out, int out_size, void* d_ws, size_t ws_size,
                              hipStream_t stream) {
  (void)in_sizes; (void)n_in; (void)out_size; (void)ws_size;
  const float* x      = (const float*)d_in[0];
  const float* W_qkv  = (const float*)d_in[1];
  const float* W_proj = (const float*)d_in[2];
  const float* b_proj = (const float*)d_in[3];
  const float* W_k    = (const float*)d_in[4];
  const float* W_gt   = (const float*)d_in[5];
  float* out = (float*)d_out;

  size_t off = 0;
  char* wsb = (char*)d_ws;
  auto take = [&](size_t bytes) -> void* {
    void* p = wsb + off;
    off += (bytes + 255) & ~(size_t)255;
    return p;
  };
  bf16_t* Y       = (bf16_t*)take((size_t)BATCH * NTOK * DIM * 2);
  bf16_t* Wqkvp   = (bf16_t*)take((size_t)DIM * 3 * DIM * 2);   // packed
  bf16_t* Wkp     = (bf16_t*)take((size_t)DIM * DIM * 2);       // packed
  bf16_t* Wprojp  = (bf16_t*)take((size_t)DIM * DIM * 2);       // packed
  float*  tvec    = (float*)take((size_t)BATCH * DIM * 4);
  float*  score   = (float*)take((size_t)BATCH * NTOK * 4);
  int*    reserve = (int*)take((size_t)BATCH * KEEP * 4);
  bf16_t* metric  = (bf16_t*)take((size_t)BATCH * KEEP * DIM * 2);
  float*  maxsim  = (float*)take((size_t)BATCH * HALF * 4);
  int*    dstall  = (int*)take((size_t)BATCH * HALF * 4);
  int*    srcidx  = (int*)take((size_t)BATCH * RNUM * 4);
  int*    unmidx  = (int*)take((size_t)BATCH * UNMN * 4);
  int*    dstidx  = (int*)take((size_t)BATCH * RNUM * 4);
  int*    invrow  = (int*)take((size_t)BATCH * HALF * 4);
  int*    counts  = (int*)take((size_t)BATCH * HALF * 4);
  bf16_t* xmerge  = (bf16_t*)take((size_t)BATCH * MN * DIM * 2);
  bf16_t* qkvb    = (bf16_t*)take((size_t)BATCH * MN * 3 * DIM * 2);
  bf16_t* vtbuf   = (bf16_t*)take((size_t)BATCH * DIM * NPAD * 2);
  bf16_t* xattn   = (bf16_t*)take((size_t)BATCH * MN * DIM * 2);

  // 1) bf16 conversion of x + packed weights
  {
    long long nx = (long long)BATCH * NTOK * DIM;
    cvt_f32_bf16<<<(unsigned)((nx + 255) / 256), 256, 0, stream>>>(x, Y, nx);
    long long nq = (long long)DIM * 3 * DIM;
    pack_b_kernel<<<(unsigned)((nq + 255) / 256), 256, 0, stream>>>(W_qkv, Wqkvp, DIM, 3 * DIM);
    long long nk = (long long)DIM * DIM;
    pack_b_kernel<<<(unsigned)((nk + 255) / 256), 256, 0, stream>>>(W_k, Wkp, DIM, DIM);
    pack_b_kernel<<<(unsigned)((nk + 255) / 256), 256, 0, stream>>>(W_proj, Wprojp, DIM, DIM);
  }
  // 2) gt / tvec
  gt_tvec_kernel<<<BATCH, DIM, 0, stream>>>(x, W_gt, W_k, tvec);
  // 3) scores (softmax skipped: monotone under argsort)
  score_kernel<<<(BATCH * NTOK) / 256, 256, 0, stream>>>(x, tvec, score);
  // 4) sorted keep order
  sort_reserve_kernel<<<BATCH, 1024, 0, stream>>>(score, reserve);
  // 5) k_res = Y[reserve] @ W_k  (gather WMMA GEMM, bf16 out)
  gemm_bf16_wmma<<<dim3(KEEP / 16, DIM / 64, BATCH), 32, 0, stream>>>(
      Y, (long long)NTOK * DIM, reserve, KEEP, Wkp, nullptr,
      nullptr, metric, (long long)KEEP * DIM, KEEP, DIM, DIM);
  // 5b) normalize rows
  normalize_metric<<<dim3(KEEP, BATCH), DIM, 0, stream>>>(metric);
  // 6) sim row max/argmax
  sim_rowmax<<<dim3(HALF / 16, BATCH), 32, 0, stream>>>(metric, maxsim, dstall);
  // 7) rank sort + index maps + counts
  sort_rank_kernel<<<BATCH, 256, 0, stream>>>(maxsim, dstall, srcidx, unmidx, dstidx, invrow, counts);
  // 8) x_merge
  build_xmerge<<<dim3(MN, BATCH), DIM, 0, stream>>>(x, reserve, unmidx, srcidx, dstidx, counts, xmerge);
  // 9) qkv = x_merge @ W_qkv (WMMA, bf16 out)
  gemm_bf16_wmma<<<dim3(MTILES, (3 * DIM) / 64, BATCH), 32, 0, stream>>>(
      xmerge, (long long)MN * DIM, nullptr, 0, Wqkvp, nullptr,
      nullptr, qkvb, (long long)MN * 3 * DIM, MN, 3 * DIM, DIM);
  // 9b) V transpose (zero-padded K-contiguous layout)
  {
    long long nv = (long long)BATCH * DIM * NPAD;
    transpose_v<<<(unsigned)((nv + 255) / 256), 256, 0, stream>>>(qkvb, vtbuf);
  }
  // 10) attention with top-k mask
  {
    size_t smem = (size_t)16 * NPAD * 4 + (size_t)16 * NPAD * 2;
    attention_kernel<<<dim3(MTILES, NHEADS, BATCH), 32, smem, stream>>>(qkvb, vtbuf, xattn);
  }
  // 11) scatter restored rows over Y (pruned tokens keep original x)
  scatter_rows<<<dim3(KEEP, BATCH), DIM, 0, stream>>>(xattn, reserve, invrow, Y);
  // 12) out = Y @ W_proj + b_proj (WMMA, f32 out)
  gemm_bf16_wmma<<<dim3(NTOK / 16, DIM / 64, BATCH), 32, 0, stream>>>(
      Y, (long long)NTOK * DIM, nullptr, 0, Wprojp, b_proj,
      out, nullptr, (long long)NTOK * DIM, NTOK, DIM, DIM);
}
